// PTDQwen2ForCausalLM_34127810134503
// MI455X (gfx1250) — matrix operand order repaired and checked by reading, not compiled, not run on hardware
//
#include <hip/hip_runtime.h>
#include <cstdint>
#include <cstddef>

// ---------------- model constants ----------------
constexpr int B_   = 2;
constexpr int S_   = 4096;
constexpr int D_   = 1024;
constexpr int H_   = 16;
constexpr int KH_  = 4;
constexpr int HD_  = 64;
constexpr int FF_  = 2816;
constexpr int L_   = 12;
constexpr int BPL_ = 6;
constexpr int SEG_ = 16;
constexpr int NSEG = 256;           // S/SEG
constexpr int KSEG = 76;            // int(NSEG*0.3)
constexpr int T_   = KSEG * SEG_;   // 1216 gathered tokens per batch
constexpr int RANK_ = 16;
constexpr int NQ_  = 8;
constexpr int MTOT = B_ * T_;       // 2432 GEMM rows

typedef unsigned short u16;
typedef __attribute__((ext_vector_type(16))) __bf16        v16bf;
typedef __attribute__((ext_vector_type(8)))  float         v8f;
typedef __attribute__((ext_vector_type(4)))  unsigned int  uint4v;
typedef __attribute__((ext_vector_type(4)))  float         f32x4;

union Frag16 { v16bf v; uint4v q[2]; };

__device__ __forceinline__ u16 f2bf(float f) {
    unsigned u = __float_as_uint(f);
    unsigned r = u + 0x7FFFu + ((u >> 16) & 1u);   // RNE
    return (u16)(r >> 16);
}

__device__ __forceinline__ v8f wmma_bf16(const Frag16& a, const Frag16& b, v8f c) {
    return __builtin_amdgcn_wmma_f32_16x16x32_bf16(false, a.v, false, b.v,
                                                   (short)0, c, false, false);
}

// ---------------- weight convert + transpose: WT[n*K+k] = bf16(W[k*N+n]) ----------------
__global__ __launch_bounds__(256) void k_convT(const float* __restrict__ W,
                                               u16* __restrict__ WT, int K, int N) {
    size_t lay = blockIdx.y;
    W  += lay * (size_t)K * N;
    WT += lay * (size_t)K * N;
    size_t idx = (size_t)blockIdx.x * 256 + threadIdx.x;
    if (idx >= (size_t)K * N) return;
    int n = (int)(idx / K), k = (int)(idx % K);
    WT[idx] = f2bf(W[(size_t)k * N + n]);
}

// ---------------- embedding gather ----------------
__global__ __launch_bounds__(256) void k_embed(const int* __restrict__ ids,
                                               const float* __restrict__ emb,
                                               float* __restrict__ hidden) {
    int bs = blockIdx.x;
    int id = ids[bs];
    const f32x4* src = (const f32x4*)(emb + (size_t)id * D_);
    f32x4* dst = (f32x4*)(hidden + (size_t)bs * D_);
    dst[threadIdx.x] = src[threadIdx.x];
}

// ---------------- segment pool + low-rank router scores ----------------
__global__ __launch_bounds__(256) void k_router(const float* __restrict__ hidden,
                                                const float* __restrict__ rk,   // [D][RANK]
                                                const float* __restrict__ rq,   // [NQ][RANK]
                                                float* __restrict__ scores) {   // [B][NSEG]
    int seg = blockIdx.x, b = blockIdx.y, tid = threadIdx.x;
    const float* base = hidden + ((size_t)b * S_ + (size_t)seg * SEG_) * D_;
    float part[RANK_];
    for (int r = 0; r < RANK_; ++r) part[r] = 0.f;
    for (int d = tid; d < D_; d += 256) {
        float pooled = 0.f;
        for (int t = 0; t < SEG_; ++t) pooled += base[(size_t)t * D_ + d];
        pooled *= (1.0f / (float)SEG_);
        for (int r = 0; r < RANK_; ++r) part[r] += pooled * rk[(size_t)d * RANK_ + r];
    }
    __shared__ float sh[256];
    __shared__ float keys[RANK_];
    for (int r = 0; r < RANK_; ++r) {
        sh[tid] = part[r]; __syncthreads();
        for (int s = 128; s > 0; s >>= 1) { if (tid < s) sh[tid] += sh[tid + s]; __syncthreads(); }
        if (tid == 0) keys[r] = sh[0];
        __syncthreads();
    }
    if (tid == 0) {
        float best = -3.0e38f;
        for (int q = 0; q < NQ_; ++q) {
            float d = 0.f;
            for (int r = 0; r < RANK_; ++r) d += rq[q * RANK_ + r] * keys[r];
            best = fmaxf(best, d);
        }
        scores[b * NSEG + seg] = best;
    }
}

// ---------------- deterministic top-KSEG with sorted indices ----------------
__global__ __launch_bounds__(NSEG) void k_topk(const float* __restrict__ scores,
                                               int* __restrict__ seg_ix) {
    int b = blockIdx.x, i = threadIdx.x;
    __shared__ float s[NSEG];
    __shared__ int sel[NSEG];
    s[i] = scores[b * NSEG + i];
    __syncthreads();
    float mine = s[i];
    int rank = 0;
    for (int j = 0; j < NSEG; ++j) {
        float o = s[j];
        rank += (o > mine) || (o == mine && j < i);
    }
    sel[i] = (rank < KSEG) ? 1 : 0;
    __syncthreads();
    if (sel[i]) {
        int pos = 0;
        for (int j = 0; j < i; ++j) pos += sel[j];
        seg_ix[b * KSEG + pos] = i;
    }
}

// ---------------- gather / scatter selected tokens ----------------
__global__ __launch_bounds__(256) void k_gather(const float* __restrict__ hidden,
                                                const int* __restrict__ seg_ix,
                                                float* __restrict__ x_sp,
                                                int* __restrict__ p_sp) {
    int bt = blockIdx.x, b = bt / T_, t = bt % T_;
    int tok = seg_ix[b * KSEG + t / SEG_] * SEG_ + (t & (SEG_ - 1));
    if (threadIdx.x == 0) p_sp[bt] = tok;
    const f32x4* src = (const f32x4*)(hidden + ((size_t)b * S_ + tok) * D_);
    f32x4* dst = (f32x4*)(x_sp + (size_t)bt * D_);
    dst[threadIdx.x] = src[threadIdx.x];
}

__global__ __launch_bounds__(256) void k_scatter(const float* __restrict__ x_sp,
                                                 const int* __restrict__ seg_ix,
                                                 float* __restrict__ hidden) {
    int bt = blockIdx.x, b = bt / T_, t = bt % T_;
    int tok = seg_ix[b * KSEG + t / SEG_] * SEG_ + (t & (SEG_ - 1));
    const f32x4* src = (const f32x4*)(x_sp + (size_t)bt * D_);
    f32x4* dst = (f32x4*)(hidden + ((size_t)b * S_ + tok) * D_);
    dst[threadIdx.x] = src[threadIdx.x];
}

// ---------------- RMSNorm (f32 in -> bf16 out) ----------------
__global__ __launch_bounds__(256) void k_rmsnorm(const float* __restrict__ X,
                                                 const float* __restrict__ w,
                                                 u16* __restrict__ out) {
    int m = blockIdx.x, tid = threadIdx.x;
    const float* x = X + (size_t)m * D_;
    float ss = 0.f;
    for (int i = tid; i < D_; i += 256) { float v = x[i]; ss += v * v; }
    __shared__ float sh[256];
    sh[tid] = ss; __syncthreads();
    for (int s = 128; s > 0; s >>= 1) { if (tid < s) sh[tid] += sh[tid + s]; __syncthreads(); }
    float r = rsqrtf(sh[0] / (float)D_ + 1e-6f);
    u16* o = out + (size_t)m * D_;
    for (int i = tid; i < D_; i += 256) o[i] = f2bf(x[i] * r * w[i]);
}

// ---------------- 128x128 WMMA GEMM, double-buffered ASYNC global->LDS pipeline ----------------
// C[M,N](f32) (+bias | +=) = A[M,K]bf16 * BT[N,K]bf16
template <bool BIAS, bool RESID>
__global__ __launch_bounds__(256) void k_gemm128(const u16* __restrict__ A,
                                                 const u16* __restrict__ BT,
                                                 const float* __restrict__ bias,
                                                 float* __restrict__ C,
                                                 int M, int N, int K) {
    __shared__ __align__(16) u16 As[2][128 * 32];
    __shared__ __align__(16) u16 Bs[2][128 * 32];
    const int tid = threadIdx.x, lane = tid & 31, wave = tid >> 5;
    const int m0 = blockIdx.y * 128, n0 = blockIdx.x * 128;
    const int wm = wave & 3;        // 4 M-subtiles of 32 rows
    const int wn = wave >> 2;       // 2 N-subtiles of 64 cols
    v8f acc[2][4];
    v8f zero = {0.f, 0.f, 0.f, 0.f, 0.f, 0.f, 0.f, 0.f};
    for (int i = 0; i < 2; ++i) for (int j = 0; j < 4; ++j) acc[i][j] = zero;

    const int lm   = lane & 15;
    const int aOff = (lane < 16) ? 0 : 16;   // A: K 0-15 / 16-31 per half-wave
    const int bOff = (lane < 16) ? 0 : 16;   // B: K 0-15 / 16-31 per half-wave
    const int lr = tid >> 1, lh = tid & 1;   // staging: 32 bytes per thread

    const unsigned laddrA[2] = { (unsigned)(size_t)&As[0][lr * 32 + lh * 16],
                                 (unsigned)(size_t)&As[1][lr * 32 + lh * 16] };
    const unsigned laddrB[2] = { (unsigned)(size_t)&Bs[0][lr * 32 + lh * 16],
                                 (unsigned)(size_t)&Bs[1][lr * 32 + lh * 16] };
    const u16* gA = A  + (size_t)(m0 + lr) * K + lh * 16;
    const u16* gB = BT + (size_t)(n0 + lr) * K + lh * 16;

    const int nsteps = K >> 5;
    // prologue: stage tile 0 into buffer 0 (ASYNCcnt-tracked copies, no VGPR round-trip)
    asm volatile("global_load_async_to_lds_b128 %0, %1, off"            :: "v"(laddrA[0]), "v"(gA) : "memory");
    asm volatile("global_load_async_to_lds_b128 %0, %1, off offset:16"  :: "v"(laddrA[0]), "v"(gA) : "memory");
    asm volatile("global_load_async_to_lds_b128 %0, %1, off"            :: "v"(laddrB[0]), "v"(gB) : "memory");
    asm volatile("global_load_async_to_lds_b128 %0, %1, off offset:16"  :: "v"(laddrB[0]), "v"(gB) : "memory");

    #pragma unroll 2
    for (int k = 0; k < nsteps; ++k) {
        const int cur = k & 1;
        // everyone finished reading buf[cur^1] (previous iteration) -> safe to prefetch into it
        __syncthreads();
        if (k + 1 < nsteps) {
            const u16* na = gA + ((k + 1) << 5);
            const u16* nb = gB + ((k + 1) << 5);
            asm volatile("global_load_async_to_lds_b128 %0, %1, off"           :: "v"(laddrA[cur ^ 1]), "v"(na) : "memory");
            asm volatile("global_load_async_to_lds_b128 %0, %1, off offset:16" :: "v"(laddrA[cur ^ 1]), "v"(na) : "memory");
            asm volatile("global_load_async_to_lds_b128 %0, %1, off"           :: "v"(laddrB[cur ^ 1]), "v"(nb) : "memory");
            asm volatile("global_load_async_to_lds_b128 %0, %1, off offset:16" :: "v"(laddrB[cur ^ 1]), "v"(nb) : "memory");
            // ASYNCcnt decrements in order: <=4 outstanding => current tile's 4 copies landed,
            // prefetch for the next tile stays in flight during compute.
            asm volatile("s_wait_asynccnt 0x4" ::: "memory");
        } else {
            asm volatile("s_wait_asynccnt 0x0" ::: "memory");
        }
        __syncthreads();   // all waves' copies into buf[cur] visible

        Frag16 af[2], bf[4];
        for (int i = 0; i < 2; ++i) {
            const u16* p = &As[cur][(wm * 32 + i * 16 + lm) * 32 + aOff];
            af[i].q[0] = *(const uint4v*)p;
            af[i].q[1] = *(const uint4v*)(p + 8);
        }
        for (int j = 0; j < 4; ++j) {
            const u16* p = &Bs[cur][(wn * 64 + j * 16 + lm) * 32 + bOff];
            bf[j].q[0] = *(const uint4v*)p;
            bf[j].q[1] = *(const uint4v*)(p + 8);
        }
        for (int i = 0; i < 2; ++i)
            for (int j = 0; j < 4; ++j)
                acc[i][j] = wmma_bf16(af[i], bf[j], acc[i][j]);
    }

    const int rAdd = (lane < 16) ? 0 : 8;
    for (int i = 0; i < 2; ++i)
        for (int j = 0; j < 4; ++j) {
            int col = n0 + wn * 64 + j * 16 + lm;
            float badd = BIAS ? bias[col] : 0.0f;
            for (int r = 0; r < 8; ++r) {
                int row = m0 + wm * 32 + i * 16 + rAdd + r;
                float v = acc[i][j][r] + badd;
                size_t off = (size_t)row * N + col;
                if (RESID) C[off] += v; else C[off] = v;
            }
        }
}

// ---------------- fused flash attention (WMMA QK^T + online softmax + WMMA PV) ----------------
// Q  : [B*H][T][64] bf16 (rope'd)
// Kt : [B*KH][T][64] bf16 (rope'd; row-major = B^T layout for QK^T)
// Vt : [B*KH][64][T] bf16 (transposed = B^T layout for PV)
// O  : token-major [B*T][H*64] bf16
__global__ __launch_bounds__(128) void k_flash(const u16* __restrict__ Q,
                                               const u16* __restrict__ Kt,
                                               const u16* __restrict__ Vt,
                                               u16* __restrict__ O) {
    const int tid = threadIdx.x, lane = tid & 31, w = tid >> 5;
    const int z = blockIdx.y;                 // b*H + h
    const int b = z >> 4, h = z & 15;
    const int zb = b * KH_ + (h >> 2);        // GQA kv batch
    const u16* Qb = Q  + (size_t)z  * T_ * HD_;
    const u16* Kb = Kt + (size_t)zb * T_ * HD_;
    const u16* Vb = Vt + (size_t)zb * HD_ * T_;
    const int m0 = blockIdx.x * 64 + w * 16;  // this wave's 16 query rows
    const int lm = lane & 15;
    const int hi = lane >> 4;                 // half-wave id
    const int rAdd = hi * 8;

    v8f zero = {0.f, 0.f, 0.f, 0.f, 0.f, 0.f, 0.f, 0.f};

    // Q fragments (A-layout), two 32-wide K-dim steps over HD=64
    Frag16 qa[2];
    {
        const int row = m0 + lm;
        const int cb = hi * 16;
        for (int ds = 0; ds < 2; ++ds) {
            const u16* p = Qb + (size_t)row * HD_ + ds * 32 + cb;
            qa[ds].q[0] = *(const uint4v*)p;
            qa[ds].q[1] = *(const uint4v*)(p + 8);
        }
    }

    v8f oacc[4];
    for (int d = 0; d < 4; ++d) oacc[d] = zero;
    float mrow[8], srow[8];
    for (int r = 0; r < 8; ++r) { mrow[r] = -3.0e38f; srow[r] = 0.f; }

    __shared__ __align__(16) u16 Pls[4][16 * 32];   // per-wave P tile
    u16* Pw = Pls[w];

    const int rowHi = m0 + 15;
    for (int j0 = 0; j0 <= rowHi; j0 += 32) {
        // ---- S = Q K^T over 32 keys (two 16-key sub-tiles) ----
        v8f s0 = zero, s1 = zero;
        for (int ds = 0; ds < 2; ++ds) {
            Frag16 kf0, kf1;
            const u16* p0 = Kb + (size_t)(j0 + lm) * HD_ + ds * 32 + hi * 16;
            kf0.q[0] = *(const uint4v*)p0; kf0.q[1] = *(const uint4v*)(p0 + 8);
            const u16* p1 = Kb + (size_t)(j0 + 16 + lm) * HD_ + ds * 32 + hi * 16;
            kf1.q[0] = *(const uint4v*)p1; kf1.q[1] = *(const uint4v*)(p1 + 8);
            s0 = wmma_bf16(qa[ds], kf0, s0);
            s1 = wmma_bf16(qa[ds], kf1, s1);
        }
        // ---- online softmax over the 16x32 tile ----
        #pragma unroll
        for (int r = 0; r < 8; ++r) {
            const int row = m0 + rAdd + r;
            const int c0 = j0 + lm, c1 = j0 + 16 + lm;
            float sv0 = (c0 <= row) ? s0[r] * 0.125f : -3.0e38f;
            float sv1 = (c1 <= row) ? s1[r] * 0.125f : -3.0e38f;
            float mx = fmaxf(sv0, sv1);
            for (int msk = 1; msk < 16; msk <<= 1)
                mx = fmaxf(mx, __shfl_xor(mx, msk, 32));
            float mnew = fmaxf(mrow[r], mx);
            float corr = expf(mrow[r] - mnew);
            mrow[r] = mnew;
            float p0 = expf(sv0 - mnew);
            float p1 = expf(sv1 - mnew);
            float psum = p0 + p1;
            for (int msk = 1; msk < 16; msk <<= 1)
                psum += __shfl_xor(psum, msk, 32);
            srow[r] = srow[r] * corr + psum;
            #pragma unroll
            for (int d = 0; d < 4; ++d) oacc[d][r] *= corr;
            Pw[(rAdd + r) * 32 + lm]      = f2bf(p0);
            Pw[(rAdd + r) * 32 + 16 + lm] = f2bf(p1);
        }
        // ---- re-layout P tile (C-layout -> A-fragment) via wave-private LDS ----
        Frag16 pf;
        {
            const u16* p = Pw + lm * 32 + hi * 16;
            pf.q[0] = *(const uint4v*)p;
            pf.q[1] = *(const uint4v*)(p + 8);
        }
        // ---- O += P (16x32) @ V (32 keys x 64 d), 4 d-blocks ----
        #pragma unroll
        for (int d = 0; d < 4; ++d) {
            Frag16 vf2;
            const u16* p = Vb + (size_t)(d * 16 + lm) * T_ + j0 + hi * 16;
            vf2.q[0] = *(const uint4v*)p; vf2.q[1] = *(const uint4v*)(p + 8);
            oacc[d] = wmma_bf16(pf, vf2, oacc[d]);
        }
    }
    // ---- normalize and write token-major bf16 output ----
    #pragma unroll
    for (int r = 0; r < 8; ++r) {
        const int t = m0 + rAdd + r;
        const float inv = 1.0f / srow[r];
        u16* dst = O + ((size_t)(b * T_ + t) * (H_ * HD_)) + h * HD_;
        dst[lm]      = f2bf(oacc[0][r] * inv);
        dst[16 + lm] = f2bf(oacc[1][r] * inv);
        dst[32 + lm] = f2bf(oacc[2][r] * inv);
        dst[48 + lm] = f2bf(oacc[3][r] * inv);
    }
}

// ---------------- RoPE + head-major reshape (Q) ----------------
__global__ __launch_bounds__(512) void k_rope_q(const float* __restrict__ qf,
                                                const int* __restrict__ p_sp,
                                                u16* __restrict__ qb) { // [B][H][T][64]
    int bt = blockIdx.x, b = bt / T_, t = bt % T_;
    int h = threadIdx.x >> 5, i = threadIdx.x & 31;
    float pos = (float)p_sp[bt];
    float inv = powf(10000.0f, -(float)(2 * i) / (float)HD_);
    float f = pos * inv, c = cosf(f), s = sinf(f);
    const float* src = qf + (size_t)bt * (H_ * HD_) + h * HD_;
    float x1 = src[i], x2 = src[i + 32];
    u16* dst = qb + (((size_t)(b * H_ + h) * T_) + t) * HD_;
    dst[i]      = f2bf(x1 * c - x2 * s);
    dst[i + 32] = f2bf(x2 * c + x1 * s);
}

__global__ __launch_bounds__(128) void k_rope_k(const float* __restrict__ kf,
                                                const int* __restrict__ p_sp,
                                                u16* __restrict__ kb) { // [B][KH][T][64]
    int bt = blockIdx.x, b = bt / T_, t = bt % T_;
    int h = threadIdx.x >> 5, i = threadIdx.x & 31;
    float pos = (float)p_sp[bt];
    float inv = powf(10000.0f, -(float)(2 * i) / (float)HD_);
    float f = pos * inv, c = cosf(f), s = sinf(f);
    const float* src = kf + (size_t)bt * (KH_ * HD_) + h * HD_;
    float x1 = src[i], x2 = src[i + 32];
    u16* dst = kb + (((size_t)(b * KH_ + h) * T_) + t) * HD_;
    dst[i]      = f2bf(x1 * c - x2 * s);
    dst[i + 32] = f2bf(x2 * c + x1 * s);
}

// ---------------- V reshape to transposed bf16 [B][KH][64][T] ----------------
__global__ __launch_bounds__(256) void k_v_reshape(const float* __restrict__ vf,
                                                   u16* __restrict__ vT) {
    int bt = blockIdx.x, b = bt / T_, t = bt % T_;
    int idx = threadIdx.x;              // KH*HD = 256
    int kh = idx >> 6, d = idx & 63;
    vT[(((size_t)(b * KH_ + kh) * HD_) + d) * T_ + t] = f2bf(vf[(size_t)bt * (KH_ * HD_) + idx]);
}

// ---------------- silu(gate)*up -> bf16 ----------------
__global__ __launch_bounds__(256) void k_silu_mul(const float* __restrict__ g,
                                                  const float* __restrict__ u,
                                                  u16* __restrict__ out, size_t n) {
    size_t i = (size_t)blockIdx.x * 256 + threadIdx.x;
    if (i >= n) return;
    float x = g[i];
    float s = x / (1.0f + expf(-x));
    out[i] = f2bf(s * u[i]);
}

// ======================= host orchestration =======================
extern "C" void kernel_launch(void* const* d_in, const int* in_sizes, int n_in,
                              void* d_out, int out_size, void* d_ws, size_t ws_size,
                              hipStream_t stream) {
    const int*   input_ids = (const int*)d_in[0];
    const float* embed     = (const float*)d_in[1];
    const float* router_k  = (const float*)d_in[2];
    const float* router_q  = (const float*)d_in[3];
    const float* ln1       = (const float*)d_in[4];
    const float* ln2       = (const float*)d_in[5];
    const float* wq        = (const float*)d_in[6];
    const float* bq        = (const float*)d_in[7];
    const float* wk        = (const float*)d_in[8];
    const float* bk        = (const float*)d_in[9];
    const float* wv        = (const float*)d_in[10];
    const float* bv        = (const float*)d_in[11];
    const float* wo        = (const float*)d_in[12];
    const float* wg        = (const float*)d_in[13];
    const float* wu        = (const float*)d_in[14];
    const float* wd        = (const float*)d_in[15];
    float* hidden = (float*)d_out;

    uint8_t* wp = (uint8_t*)d_ws;
    auto take = [&](size_t bytes) -> void* {
        void* r = (void*)wp;
        wp += (bytes + 255) & ~(size_t)255;
        return r;
    };
    u16* wqT = (u16*)take((size_t)L_ * D_ * (H_ * HD_) * 2);
    u16* wkT = (u16*)take((size_t)L_ * D_ * (KH_ * HD_) * 2);
    u16* wvT = (u16*)take((size_t)L_ * D_ * (KH_ * HD_) * 2);
    u16* woT = (u16*)take((size_t)L_ * (H_ * HD_) * D_ * 2);
    u16* wgT = (u16*)take((size_t)L_ * D_ * FF_ * 2);
    u16* wuT = (u16*)take((size_t)L_ * D_ * FF_ * 2);
    u16* wdT = (u16*)take((size_t)L_ * FF_ * D_ * 2);
    float* segsc = (float*)take((size_t)B_ * NSEG * 4);
    int*   segix = (int*)take((size_t)B_ * KSEG * 4);
    float* x_sp  = (float*)take((size_t)MTOT * D_ * 4);
    int*   p_sp  = (int*)take((size_t)MTOT * 4);
    u16*   hb    = (u16*)take((size_t)MTOT * D_ * 2);
    u16*   h2b   = (u16*)take((size_t)MTOT * D_ * 2);
    u16*   ob    = (u16*)take((size_t)MTOT * (H_ * HD_) * 2);
    float* qf    = (float*)take((size_t)MTOT * (H_ * HD_) * 4);
    float* kf    = (float*)take((size_t)MTOT * (KH_ * HD_) * 4);
    float* vf    = (float*)take((size_t)MTOT * (KH_ * HD_) * 4);
    u16*   qbf   = (u16*)take((size_t)B_ * H_ * T_ * HD_ * 2);
    u16*   kbf   = (u16*)take((size_t)B_ * KH_ * T_ * HD_ * 2);
    u16*   vTbf  = (u16*)take((size_t)B_ * KH_ * HD_ * T_ * 2);
    float* gate  = (float*)take((size_t)MTOT * FF_ * 4);
    float* up    = (float*)take((size_t)MTOT * FF_ * 4);
    u16*   ffb   = (u16*)take((size_t)MTOT * FF_ * 2);

    auto cgrid = [](int K, int N) {
        return dim3((unsigned)(((size_t)K * N + 255) / 256), (unsigned)L_, 1);
    };
    // one-time bf16 transposed weights
    k_convT<<<cgrid(D_, H_ * HD_), 256, 0, stream>>>(wq, wqT, D_, H_ * HD_);
    k_convT<<<cgrid(D_, KH_ * HD_), 256, 0, stream>>>(wk, wkT, D_, KH_ * HD_);
    k_convT<<<cgrid(D_, KH_ * HD_), 256, 0, stream>>>(wv, wvT, D_, KH_ * HD_);
    k_convT<<<cgrid(H_ * HD_, D_), 256, 0, stream>>>(wo, woT, H_ * HD_, D_);
    k_convT<<<cgrid(D_, FF_), 256, 0, stream>>>(wg, wgT, D_, FF_);
    k_convT<<<cgrid(D_, FF_), 256, 0, stream>>>(wu, wuT, D_, FF_);
    k_convT<<<cgrid(FF_, D_), 256, 0, stream>>>(wd, wdT, FF_, D_);

    k_embed<<<B_ * S_, 256, 0, stream>>>(input_ids, embed, hidden);

    for (int blk = 0; blk < 2; ++blk) {
        k_router<<<dim3(NSEG, B_), 256, 0, stream>>>(
            hidden, router_k + (size_t)blk * D_ * RANK_,
            router_q + (size_t)blk * NQ_ * RANK_, segsc);
        k_topk<<<B_, NSEG, 0, stream>>>(segsc, segix);
        k_gather<<<MTOT, 256, 0, stream>>>(hidden, segix, x_sp, p_sp);

        for (int j = 0; j < BPL_; ++j) {
            int li = blk * BPL_ + j;
            // --- attention ---
            k_rmsnorm<<<MTOT, 256, 0, stream>>>(x_sp, ln1 + (size_t)li * D_, hb);
            k_gemm128<true, false><<<dim3(8, 19), 256, 0, stream>>>(
                hb, wqT + (size_t)li * D_ * 1024, bq + (size_t)li * 1024, qf, MTOT, 1024, D_);
            k_gemm128<true, false><<<dim3(4, 19), 256, 0, stream>>>(
                hb, wkT + (size_t)li * D_ * 512, bk + (size_t)li * 512, kf, MTOT, 512, D_);
            k_gemm128<true, false><<<dim3(4, 19), 256, 0, stream>>>(
                hb, wvT + (size_t)li * D_ * 512, bv + (size_t)li * 512, vf, MTOT, 512, D_);
            k_rope_q<<<MTOT, 512, 0, stream>>>(qf, p_sp, qbf);
            k_rope_k<<<MTOT, 128, 0, stream>>>(kf, p_sp, kbf);
            k_v_reshape<<<MTOT, 256, 0, stream>>>(vf, vTbf);
            // fused flash attention -> ob (token-major bf16)
            k_flash<<<dim3(T_ / 64, B_ * H_), 128, 0, stream>>>(qbf, kbf, vTbf, ob);
            k_gemm128<false, true><<<dim3(8, 19), 256, 0, stream>>>(
                ob, woT + (size_t)li * 1024 * D_, nullptr, x_sp, MTOT, D_, 1024);
            // --- MLP ---
            k_rmsnorm<<<MTOT, 256, 0, stream>>>(x_sp, ln2 + (size_t)li * D_, h2b);
            k_gemm128<false, false><<<dim3(22, 19), 256, 0, stream>>>(
                h2b, wgT + (size_t)li * D_ * FF_, nullptr, gate, MTOT, FF_, D_);
            k_gemm128<false, false><<<dim3(22, 19), 256, 0, stream>>>(
                h2b, wuT + (size_t)li * D_ * FF_, nullptr, up, MTOT, FF_, D_);
            k_silu_mul<<<(unsigned)(((size_t)MTOT * FF_ + 255) / 256), 256, 0, stream>>>(
                gate, up, ffb, (size_t)MTOT * FF_);
            k_gemm128<false, true><<<dim3(8, 19), 256, 0, stream>>>(
                ffb, wdT + (size_t)li * FF_ * D_, nullptr, x_sp, MTOT, D_, FF_);
        }
        k_scatter<<<MTOT, 256, 0, stream>>>(x_sp, segix, hidden);
    }
}
